// HolEInteraction_9517647528092
// MI455X (gfx1250) — matrix operand order repaired and checked by reading, compiled, not run on hardware
//
#include <hip/hip_runtime.h>

typedef __attribute__((ext_vector_type(16))) _Float16 v16h;
typedef __attribute__((ext_vector_type(8)))  _Float16 v8h;
typedef __attribute__((ext_vector_type(8)))  float    v8f;

#define DDIM 1024
#define WAVES 8

// out[b] = sum_{u,v} r[16u+v] * corr[16u+v],
// corr[16u+v] = sum_{j'} h[(j'-16u) mod d] * t[(j'+v) mod d]
//   A[u,K] = h[(K-16u) mod d]   (M = u, four 16-row tiles)
//   B[K,v] = t[(K+v) mod d]     (N = v, shared across tiles)
__global__ __launch_bounds__(256) void HolE_wmma_kernel(
    const float* __restrict__ h,
    const float* __restrict__ r,
    const float* __restrict__ t,
    float* __restrict__ out,
    int nrows)
{
    __shared__ __align__(16) _Float16 h_s[WAVES][2 * DDIM];    // h duplicated for wraparound
    __shared__ __align__(16) _Float16 t_s[WAVES][DDIM + 32];   // t + 32-elem tail

    const int lane = threadIdx.x & 31;
    const int w    = threadIdx.x >> 5;
    const int row  = blockIdx.x * WAVES + w;
    const bool active = (row < nrows);

    // ---- load + f32->f16 convert into LDS (coalesced float4 loads) ----
    if (active) {
        const float4* h4 = (const float4*)(h + (size_t)row * DDIM);
        const float4* t4 = (const float4*)(t + (size_t)row * DDIM);
#pragma unroll
        for (int c = 0; c < 8; ++c) {
            const int i4 = c * 32 + lane;      // float4 index 0..255
            const float4 hv = h4[i4];
            const float4 tv = t4[i4];
            const int b0 = i4 * 4;
            h_s[w][b0 + 0] = (_Float16)hv.x;  h_s[w][b0 + 1] = (_Float16)hv.y;
            h_s[w][b0 + 2] = (_Float16)hv.z;  h_s[w][b0 + 3] = (_Float16)hv.w;
            h_s[w][DDIM + b0 + 0] = (_Float16)hv.x;  h_s[w][DDIM + b0 + 1] = (_Float16)hv.y;
            h_s[w][DDIM + b0 + 2] = (_Float16)hv.z;  h_s[w][DDIM + b0 + 3] = (_Float16)hv.w;
            t_s[w][b0 + 0] = (_Float16)tv.x;  t_s[w][b0 + 1] = (_Float16)tv.y;
            t_s[w][b0 + 2] = (_Float16)tv.z;  t_s[w][b0 + 3] = (_Float16)tv.w;
            if (b0 < 32) {                     // wraparound tail t[0..31]
                t_s[w][DDIM + b0 + 0] = (_Float16)tv.x;
                t_s[w][DDIM + b0 + 1] = (_Float16)tv.y;
                t_s[w][DDIM + b0 + 2] = (_Float16)tv.z;
                t_s[w][DDIM + b0 + 3] = (_Float16)tv.w;
            }
        }
    }
    __syncthreads();
    if (!active) return;   // no barriers after this point

    const int Msel = lane & 15;        // u_local (A tiles) / v (B cols)
    const int hig  = (lane >> 4) & 1;  // lane half

    v8f acc[4] = {};                   // D[u,v] accumulators, 4 u-tiles

    // ---- 16x1024 x 1024x16(x4) matmul: 32 K-steps x 4 WMMAs ----
#pragma unroll 2
    for (int kk = 0; kk < 32; ++kk) {
        const int kb = kk * 32;

        // B fragment (t side, +v shift, element-granular addresses)
        // lanes 0-15: N=lane, K groups {0..7},{16..23}; lanes 16-31: N=lane-16, K groups {8..15},{24..31}
        v16h bfrag;
        const int tb = kb + Msel + (hig << 3);
#pragma unroll
        for (int i = 0; i < 16; ++i) bfrag[i] = t_s[w][tb + i + (i >= 8 ? 8 : 0)];

#pragma unroll
        for (int tl = 0; tl < 4; ++tl) {
            // A fragment (h side, -16u shift -> 16B-aligned b128 loads)
            // lanes 0-15: K groups {0..7},{16..23}; lanes 16-31: {8..15},{24..31}
            const int abase = DDIM + kb - 256 * tl - 16 * Msel + (hig << 3);
            const v8h alo = *(const v8h*)&h_s[w][abase];
            const v8h ahi = *(const v8h*)&h_s[w][abase + 16];
            const v16h afrag = __builtin_shufflevector(
                alo, ahi, 0, 1, 2, 3, 4, 5, 6, 7, 8, 9, 10, 11, 12, 13, 14, 15);

            acc[tl] = __builtin_amdgcn_wmma_f32_16x16x32_f16(
                false, afrag, false, bfrag, (short)0, acc[tl], false, false);
        }
    }

    // ---- dot with r + wave32 reduction ----
    // D layout: lane -> v = lane&15 ; VGPR i -> u_local = i + 8*(lane>=16)
    const float* rrow = r + (size_t)row * DDIM;
    float s = 0.0f;
#pragma unroll
    for (int tl = 0; tl < 4; ++tl) {
#pragma unroll
        for (int i = 0; i < 8; ++i) {
            const int u_local = i + (hig << 3);
            s += acc[tl][i] * rrow[256 * tl + 16 * u_local + Msel];  // coalesced
        }
    }
    s += __shfl_xor(s, 16, 32);
    s += __shfl_xor(s, 8, 32);
    s += __shfl_xor(s, 4, 32);
    s += __shfl_xor(s, 2, 32);
    s += __shfl_xor(s, 1, 32);
    if (lane == 0) out[row] = s;
}

extern "C" void kernel_launch(void* const* d_in, const int* in_sizes, int n_in,
                              void* d_out, int out_size, void* d_ws, size_t ws_size,
                              hipStream_t stream)
{
    const float* h = (const float*)d_in[0];
    const float* r = (const float*)d_in[1];
    const float* t = (const float*)d_in[2];
    float* out = (float*)d_out;

    const int nrows  = in_sizes[0] / DDIM;          // 131072
    const int blocks = (nrows + WAVES - 1) / WAVES; // 16384

    hipLaunchKernelGGL(HolE_wmma_kernel, dim3(blocks), dim3(256), 0, stream,
                       h, r, t, out, nrows);
}
